// DynamicClipAttention_77524159693557
// MI455X (gfx1250) — matrix operand order repaired and checked by reading, not compiled, not run on hardware
//
#include <hip/hip_runtime.h>
#include <stdint.h>

// ---------------------------------------------------------------------------
// DynamicClipAttention fused kernel for MI455X (gfx1250, wave32, WMMA)
//   dir 0: queries=v1, keys/values=v2  -> attended_v1
//   dir 1: queries=v2, keys/values=v1  -> attended_v2
// One workgroup = one (batch, 16-query tile, direction).
//   Phase 1: S(16x2048) = Q * K^T via v_wmma_f32_16x16x32_f16, kept in LDS
//   Phase 2: per-row max + sum(exp) (softmax denominator over ALL keys)
//   Phase 3: exact top-128 per row via 4-round radix select on float keys
//   Phase 4: weights = exp(s-m)/Z for selected, 0 otherwise (in-place)
//   Phase 5: O(16x64) = P * V via v_wmma_f32_16x16x32_f16, ds_add_f32 reduce
// ---------------------------------------------------------------------------

typedef __attribute__((ext_vector_type(16))) _Float16 v16h;
typedef __attribute__((ext_vector_type(8)))  float    v8f;

static constexpr int BATCH = 16;
static constexpr int L     = 2048;
static constexpr int D     = 64;
static constexpr int TOPK  = 128;
static constexpr int TQ    = 16;    // query rows per workgroup
static constexpr int SS    = 2056;  // padded score row stride (floats)

// LDS layout (dynamic shared memory)
static constexpr size_t OFF_QTILE = 0;                                          // TQ*D f16
static constexpr size_t OFF_OUT   = OFF_QTILE + (size_t)TQ * D * 2;             // TQ*D f32
static constexpr size_t OFF_RMAX  = OFF_OUT   + (size_t)TQ * D * 4;             // TQ f32
static constexpr size_t OFF_RZ    = OFF_RMAX  + (size_t)TQ * 4;                 // TQ f32
static constexpr size_t OFF_THR   = OFF_RZ    + (size_t)TQ * 4;                 // TQ u32
static constexpr size_t OFF_TIE   = OFF_THR   + (size_t)TQ * 4;                 // TQ i32
static constexpr size_t OFF_WANT  = OFF_TIE   + (size_t)TQ * 4;                 // TQ i32
static constexpr size_t OFF_PFX   = OFF_WANT  + (size_t)TQ * 4;                 // TQ u32
static constexpr size_t OFF_RED   = OFF_PFX   + (size_t)TQ * 4;                 // TQ*16 f32
static constexpr size_t OFF_HIST  = OFF_RED   + (size_t)TQ * 16 * 4;            // TQ*256 i32
static constexpr size_t OFF_SC    = (OFF_HIST + (size_t)TQ * 256 * 4 + 15) & ~(size_t)15;
static constexpr size_t SMEM_BYTES = OFF_SC + (size_t)TQ * SS * 4;              // ~152 KB

// 16-bit A matrix 16x32 — K index held by half-j of a lane (contiguous: k = 16*g + j)
__device__ __forceinline__ int a_k_of(int lane, int j) {
    const int g = lane >> 4;
    return 16 * g + j;
}

// Monotone float -> uint mapping (order-preserving, handles negatives)
__device__ __forceinline__ unsigned fkey(float f) {
    unsigned u = __float_as_uint(f);
    return u ^ ((u & 0x80000000u) ? 0xFFFFFFFFu : 0x80000000u);
}

__global__ void dca_kernel(const float* __restrict__ v1,
                           const unsigned char* __restrict__ v1_mask,
                           const float* __restrict__ v2,
                           const unsigned char* __restrict__ v2_mask,
                           float* __restrict__ out)
{
    const int qt   = blockIdx.x;        // query tile (0..127)
    const int b    = blockIdx.y;        // batch
    const int dir  = blockIdx.z;        // 0: v1->v2, 1: v2->v1
    const int tid  = threadIdx.x;
    const int lane = tid & 31;
    const int wave = tid >> 5;
    const int q0   = qt * TQ;
    const int r16  = lane & 15;
    const int g    = lane >> 4;

    const float* qptr  = (dir == 0 ? v1 : v2) + (size_t)b * L * D;
    const float* kvptr = (dir == 0 ? v2 : v1) + (size_t)b * L * D;
    const unsigned char* qm = (dir == 0 ? v1_mask : v2_mask) + (size_t)b * L;
    const unsigned char* km = (dir == 0 ? v2_mask : v1_mask) + (size_t)b * L;

    extern __shared__ char smem[];
    _Float16* qtile  = (_Float16*)(smem + OFF_QTILE);
    float*    outAcc = (float*)   (smem + OFF_OUT);
    float*    rowMax = (float*)   (smem + OFF_RMAX);
    float*    rowZ   = (float*)   (smem + OFF_RZ);
    unsigned* thrU   = (unsigned*)(smem + OFF_THR);
    int*      tieCnt = (int*)     (smem + OFF_TIE);
    int*      want   = (int*)     (smem + OFF_WANT);
    unsigned* pfx    = (unsigned*)(smem + OFF_PFX);
    float*    red    = (float*)   (smem + OFF_RED);
    int*      hist   = (int*)     (smem + OFF_HIST);
    float*    scores = (float*)   (smem + OFF_SC);

    // ---- Phase 0: stage Q tile (f32 -> f16) and zero the output accumulator
    for (int i = tid; i < TQ * D; i += blockDim.x) {
        qtile[i]  = (_Float16)qptr[(size_t)(q0 + (i >> 6)) * D + (i & 63)];
        outAcc[i] = 0.f;
    }
    __syncthreads();

    // ---- Phase 1: S = Q * K^T   (each wave owns key chunks of 16)
    v16h aQ[2];
#pragma unroll
    for (int f = 0; f < 2; ++f)
#pragma unroll
        for (int j = 0; j < 16; ++j)
            aQ[f][j] = qtile[r16 * D + f * 32 + a_k_of(lane, j)];

    for (int kc = wave; kc < L / 16; kc += 8) {
        const int key0 = kc * 16;
        // prefetch next chunk's key row for this lane
        if (kc + 8 < L / 16)
            __builtin_prefetch(kvptr + (size_t)(key0 + 128 + r16) * D, 0, 1);

        v8f acc = {0.f, 0.f, 0.f, 0.f, 0.f, 0.f, 0.f, 0.f};
#pragma unroll
        for (int f = 0; f < 2; ++f) {
            // B[k][n] = Key[key0+n][d], d = f*32 + 16*g + j  (contiguous per lane)
            v16h bf;
            const float* krow = kvptr + (size_t)(key0 + r16) * D + f * 32 + g * 16;
#pragma unroll
            for (int j = 0; j < 16; ++j) bf[j] = (_Float16)krow[j];
            acc = __builtin_amdgcn_wmma_f32_16x16x32_f16(
                false, aQ[f], false, bf, (short)0, acc, false, false);
        }
        // scatter C tile to LDS score strip; apply key padding mask
        const int   col   = key0 + r16;
        const bool  kmask = km[col] != 0;
#pragma unroll
        for (int i = 0; i < 8; ++i) {
            const int row = i + 8 * g;
            scores[row * SS + col] = kmask ? -3.0e38f : acc[i];
        }
    }
    __syncthreads();

    // ---- Phase 2: per-row max, then Z = sum(exp(s - m)) over ALL keys
    {
        const int row = tid >> 4, sub = tid & 15;
        float m = -3.4e38f;
        for (int c = sub; c < L; c += 16) m = fmaxf(m, scores[row * SS + c]);
        red[row * 16 + sub] = m;
    }
    __syncthreads();
    if (tid < TQ) {
        float m = red[tid * 16];
        for (int s = 1; s < 16; ++s) m = fmaxf(m, red[tid * 16 + s]);
        rowMax[tid] = m;
    }
    __syncthreads();
    {
        const int row = tid >> 4, sub = tid & 15;
        const float m = rowMax[row];
        float s = 0.f;
        for (int c = sub; c < L; c += 16) s += __expf(scores[row * SS + c] - m);
        red[row * 16 + sub] = s;
    }
    __syncthreads();
    if (tid < TQ) {
        float s = 0.f;
        for (int k = 0; k < 16; ++k) s += red[tid * 16 + k];
        rowZ[tid] = s;
        want[tid] = TOPK;
        pfx[tid]  = 0u;
    }
    __syncthreads();

    // ---- Phase 3: exact 128th-largest per row (radix select, 4 x 8-bit rounds)
    for (int round = 0; round < 4; ++round) {
        const int shift = 24 - 8 * round;
        for (int i = tid; i < TQ * 256; i += blockDim.x) hist[i] = 0;
        __syncthreads();
        {
            const int row = tid >> 4, sub = tid & 15;
            const unsigned p = pfx[row];
            for (int c = sub; c < L; c += 16) {
                const unsigned u = fkey(scores[row * SS + c]);
                const bool match =
                    (round == 0) || ((u >> (shift + 8)) == (p >> (shift + 8)));
                if (match) atomicAdd(&hist[row * 256 + ((u >> shift) & 255)], 1);
            }
        }
        __syncthreads();
        if (tid < TQ) {
            int need = want[tid];
            int cum  = 0;
            unsigned bin = 0;
            for (int bI = 255; bI >= 0; --bI) {
                const int h = hist[tid * 256 + bI];
                if (cum + h >= need) { bin = (unsigned)bI; need -= cum; break; }
                cum += h;
            }
            pfx[tid] |= bin << shift;
            want[tid] = need;  // count still needed within the chosen bin
        }
        __syncthreads();
    }
    if (tid < TQ) { thrU[tid] = pfx[tid]; tieCnt[tid] = want[tid]; }
    __syncthreads();

    // ---- Phase 4: in-place convert scores -> masked softmax weights
    {
        const int row = tid >> 4, sub = tid & 15;
        const float m    = rowMax[row];
        const float z    = rowZ[row];
        const float invZ = (z > 0.f) ? 1.0f / z : 0.f;
        const unsigned thr = thrU[row];
        for (int c = sub; c < L; c += 16) {
            const float s = scores[row * SS + c];
            const unsigned u = fkey(s);
            bool keep = (u > thr);
            if (!keep && u == thr && atomicSub(&tieCnt[row], 1) > 0) keep = true;
            scores[row * SS + c] = keep ? __expf(s - m) * invZ : 0.f;
        }
    }
    __syncthreads();

    // ---- Phase 5: O = P * V   (each wave owns key chunks of 32, LDS reduce)
    v8f oacc[4];
#pragma unroll
    for (int nb = 0; nb < 4; ++nb)
        oacc[nb] = (v8f){0.f, 0.f, 0.f, 0.f, 0.f, 0.f, 0.f, 0.f};

    for (int ks = wave; ks < L / 32; ks += 8) {
        const int keyBase = ks * 32;
        v16h aP;  // A = P chunk 16x32 (row r16, K = key offset)
#pragma unroll
        for (int j = 0; j < 16; ++j)
            aP[j] = (_Float16)scores[r16 * SS + keyBase + a_k_of(lane, j)];
#pragma unroll
        for (int nb = 0; nb < 4; ++nb) {
            v16h bV;  // B[k][n] = V[keyBase+k][nb*16+n]
            const int d = nb * 16 + r16;
#pragma unroll
            for (int j = 0; j < 16; ++j)
                bV[j] = (_Float16)kvptr[(size_t)(keyBase + 16 * g + j) * D + d];
            oacc[nb] = __builtin_amdgcn_wmma_f32_16x16x32_f16(
                false, aP, false, bV, (short)0, oacc[nb], false, false);
        }
    }
#pragma unroll
    for (int nb = 0; nb < 4; ++nb)
#pragma unroll
        for (int i = 0; i < 8; ++i)
            atomicAdd(&outAcc[(i + 8 * g) * D + nb * 16 + r16], oacc[nb][i]);
    __syncthreads();

    // ---- Phase 6: apply query mask, write flat [attended_v1 | attended_v2]
    for (int i = tid; i < TQ * D; i += blockDim.x) {
        const int r = i >> 6, d = i & 63;
        const int q = q0 + r;
        const float val = qm[q] ? 0.f : outAcc[i];
        out[(size_t)dir * BATCH * L * D + (size_t)b * L * D + (size_t)q * D + d] = val;
    }
}

extern "C" void kernel_launch(void* const* d_in, const int* in_sizes, int n_in,
                              void* d_out, int out_size, void* d_ws, size_t ws_size,
                              hipStream_t stream) {
    (void)in_sizes; (void)n_in; (void)out_size; (void)d_ws; (void)ws_size;
    const float*         v1  = (const float*)d_in[0];
    const unsigned char* v1m = (const unsigned char*)d_in[1];
    const float*         v2  = (const float*)d_in[2];
    const unsigned char* v2m = (const unsigned char*)d_in[3];
    float* out = (float*)d_out;

    // Allow >64KB dynamic LDS (WGP has 320KB); deterministic + capture-safe.
    (void)hipFuncSetAttribute(reinterpret_cast<const void*>(dca_kernel),
                              hipFuncAttributeMaxDynamicSharedMemorySize,
                              (int)SMEM_BYTES);

    dim3 grid(L / TQ, BATCH, 2);
    dca_kernel<<<grid, 256, SMEM_BYTES, stream>>>(v1, v1m, v2, v2m, out);
}
